// MultiAgentsEncoder_14328010900269
// MI455X (gfx1250) — matrix operand order repaired and checked by reading, not compiled, not run on hardware
//
#include <hip/hip_runtime.h>

// Problem constants (from reference): T=512, B=8, A=8, H=512, DIN=512, L=2, N=64
#define T_   512
#define Nn   64
#define Hh   512
#define DIN_ 512
#define G3   1536   // 3*H
#define NBLK 8      // persistent blocks for the recurrent scan (each owns 64 hidden units)

typedef __attribute__((ext_vector_type(16))) __bf16 v16bf;
typedef __attribute__((ext_vector_type(8)))  float  v8f;

union FragBF { v16bf v; unsigned int u[8]; };

__device__ __forceinline__ unsigned short f2bf(float f) {
  unsigned int x = __float_as_uint(f);
  unsigned int r = x + 0x7FFFu + ((x >> 16) & 1u);   // RNE
  return (unsigned short)(r >> 16);
}

// ---------------------------------------------------------------------------
// Tensor Data Mover support (CDNA5). Guarded so the file still compiles on a
// toolchain without the builtin (falls back to manual VMEM+DS staging).
// ---------------------------------------------------------------------------
#if defined(__has_builtin)
#if __has_builtin(__builtin_amdgcn_tensor_load_to_lds) && \
    __has_builtin(__builtin_amdgcn_s_wait_tensorcnt)
#define HAVE_TDM 1
#endif
#endif
#ifndef HAVE_TDM
#define HAVE_TDM 0
#endif

#if HAVE_TDM
typedef __attribute__((ext_vector_type(4))) unsigned int tdm_v4u;
typedef __attribute__((ext_vector_type(8))) int          tdm_v8i;
typedef __attribute__((ext_vector_type(4))) int          tdm_v4i;

// 2D tile load, bf16 elements (data_size=2B), LDS padding enabled.
// pad_interval: 0=2,1=4,...,7=256 DWORDs between pads; pad_amount: 0=1,...  DWORDs.
__device__ __forceinline__ void tdm_load_tile_2d(
    unsigned int lds_byte_addr, const void* gptr,
    unsigned int row_len,        // D#.tensor_dim0  (elements; OOB in x returns 0)
    unsigned int rows,           // D#.tensor_dim1  (rows; OOB in y returns 0)
    unsigned int row_stride,     // D#.tensor_dim0_stride (elements)
    unsigned int tile_d0, unsigned int tile_d1,
    unsigned int pad_interval, unsigned int pad_amount) {
  unsigned long long ga = (unsigned long long)gptr;
  tdm_v4u g0;
  g0.x = 1u;                                               // count=1, user mode
  g0.y = lds_byte_addr;                                    // lds_addr
  g0.z = (unsigned int)ga;                                 // global_addr[31:0]
  g0.w = (unsigned int)((ga >> 32) & 0x01FFFFFFull) | (2u << 30);  // [56:32] | type=2
  tdm_v8i g1;
  g1[0] = (int)((1u << 16)                 // data_size = 1 -> 2 bytes
              | (1u << 20)                 // pad_enable
              | (pad_interval << 22)
              | (pad_amount << 25));
  g1[1] = (int)((row_len & 0xFFFFu) << 16);                        // tensor_dim0 lo
  g1[2] = (int)(((row_len >> 16) & 0xFFFFu) | ((rows & 0xFFFFu) << 16));
  g1[3] = (int)(((rows >> 16) & 0xFFFFu) | ((tile_d0 & 0xFFFFu) << 16));
  g1[4] = (int)(tile_d1 & 0xFFFFu);                                // tile_dim1 | tile_dim2=0
  g1[5] = (int)row_stride;                                         // dim0 stride lo
  g1[6] = 0;                                                       // stride hi | dim1 stride
  g1[7] = 0;
  tdm_v4i z4 = {0, 0, 0, 0};
#if __has_include(<hip/amd_detail/amd_gfx1250_TDM.h>)
  tdm_v8i z8 = {0, 0, 0, 0, 0, 0, 0, 0};
  __builtin_amdgcn_tensor_load_to_lds(g0, g1, z4, z4, z8, 0);      // 6-arg toolchain
#else
  __builtin_amdgcn_tensor_load_to_lds(g0, g1, z4, z4, 0);          // 5-arg (ROCm 7.2)
#endif
}
#endif  // HAVE_TDM

// ---------------------------------------------------------------------------
// f32 -> bf16 conversion (grid-stride)
// ---------------------------------------------------------------------------
__global__ void mae_cvt_bf16(const float* __restrict__ s, unsigned short* __restrict__ d,
                             long long n) {
  long long i = (long long)blockIdx.x * blockDim.x + threadIdx.x;
  long long st = (long long)gridDim.x * blockDim.x;
  for (; i < n; i += st) d[i] = f2bf(s[i]);
}

// A-row remap for the backward GRU (article[::-1] without materializing it)
__device__ __forceinline__ int amap(int r, int rev) {
  if (!rev) return r;
  int t = r >> 6, n = r & 63;
  return ((T_ - 1 - t) << 6) | n;
}

// ---------------------------------------------------------------------------
// Generic WMMA bf16 GEMM:  C[M,N](f32) (+)= A[M,K] @ B[K,N] (+bias[N]) (+Drow[row%64,:])
// optional bf16 mirror of C.  Block tile 128x128, 8 waves (2x4), K-step 32.
// A tile staged via TDM (pad_interval=16DW, pad=1DW -> 34-ushort stride);
// B tile staged transposed by the waves (TDM cannot transpose).
// ---------------------------------------------------------------------------
__global__ void __launch_bounds__(256)
mae_gemm(const unsigned short* __restrict__ A, int lda,
         const unsigned short* __restrict__ B, int ldb,
         const float* __restrict__ bias,
         const float* __restrict__ Drow,
         float* __restrict__ C, unsigned short* __restrict__ Cbf,
         int M, int N, int K, int accF, int revF) {
  __shared__ unsigned short As[128 * 34];   // 34 = 32 + 1-DWORD pad per row
  __shared__ unsigned short Bs[128 * 34];   // B staged transposed [col][k]
  const int tid = threadIdx.x;
  const int lane = tid & 31, wid = tid >> 5;
  const int wm = wid & 1, wn = wid >> 1;            // wave grid 2(M) x 4(N)
  const int r0 = blockIdx.y * 128, n0 = blockIdx.x * 128;
  const int row16 = lane & 15, hi = lane >> 4;
#if HAVE_TDM
  const unsigned int as_base = (unsigned int)(size_t)(void*)As;
#endif

  v8f acc[4][2];
  const v8f vz = {0.f, 0.f, 0.f, 0.f, 0.f, 0.f, 0.f, 0.f};
  for (int i = 0; i < 4; i++) for (int j = 0; j < 2; j++) acc[i][j] = vz;

  for (int k0 = 0; k0 < K; k0 += 32) {
#if HAVE_TDM
    if (wid == 0) {
      if (!revF) {
        unsigned int rows = (unsigned int)(M - r0);
        if (rows > 128u) rows = 128u;
        tdm_load_tile_2d(as_base, A + (size_t)r0 * lda + k0,
                         (unsigned int)(lda - k0), rows, (unsigned int)lda,
                         32u, rows, 3u, 0u);
      } else {
        // reversed time: a 128-row tile is two contiguous 64-row segments
        int t0 = r0 >> 6;
        tdm_load_tile_2d(as_base, A + (size_t)((T_ - 1 - t0) * 64) * lda + k0,
                         (unsigned int)(lda - k0), 64u, (unsigned int)lda,
                         32u, 64u, 3u, 0u);
        tdm_load_tile_2d(as_base + 64u * 34u * 2u,
                         A + (size_t)((T_ - 2 - t0) * 64) * lda + k0,
                         (unsigned int)(lda - k0), 64u, (unsigned int)lda,
                         32u, 64u, 3u, 0u);
      }
    }
#else
    for (int i = 0; i < 8; i++) {
      int u = tid + 256 * i;            // 0..2047 uints
      int ar = u >> 4;                  // 0..127
      int kk = (u & 15) << 1;           // even
      int gr = r0 + ar;
      unsigned int val = 0;
      if (gr < M) {
        int sr = amap(gr, revF);
        val = *(const unsigned int*)(A + (size_t)sr * lda + k0 + kk);
      }
      *(unsigned int*)(&As[ar * 34 + kk]) = val;
    }
#endif
    // stage B tile transposed: Bs[col][k]  (overlaps with the TDM DMA)
    for (int i = 0; i < 16; i++) {
      int e = tid + 256 * i;            // 0..4095
      int col = e & 127;
      int kk = e >> 7;                  // 0..31
      Bs[col * 34 + kk] = B[(size_t)(k0 + kk) * ldb + n0 + col];
    }
#if HAVE_TDM
    if (wid == 0) __builtin_amdgcn_s_wait_tensorcnt(0);
#endif
    __syncthreads();

    FragBF af[4], bfg[2];
    for (int mt = 0; mt < 4; mt++) {
      int ar = 64 * wm + 16 * mt + row16;
      for (int v = 0; v < 4; v++) {
        int k = 2 * v + 8 * hi;
        af[mt].u[v]     = *(const unsigned int*)(&As[ar * 34 + k]);
        af[mt].u[v + 4] = *(const unsigned int*)(&As[ar * 34 + 16 + k]);
      }
    }
    for (int nt = 0; nt < 2; nt++) {
      int bc = 32 * wn + 16 * nt + row16;
      for (int v = 0; v < 8; v++) {
        int k = 8 * hi + 2 * (v & 3) + 16 * (v >> 2);
        bfg[nt].u[v] = *(const unsigned int*)(&Bs[bc * 34 + k]);
      }
    }
    for (int mt = 0; mt < 4; mt++)
      for (int nt = 0; nt < 2; nt++)
        acc[mt][nt] = __builtin_amdgcn_wmma_f32_16x16x32_bf16(
            false, af[mt].v, false, bfg[nt].v, (short)0, acc[mt][nt], false, false);
    __syncthreads();
  }

  for (int mt = 0; mt < 4; mt++) {
    for (int nt = 0; nt < 2; nt++) {
      int gcol = n0 + 32 * wn + 16 * nt + row16;
      for (int v = 0; v < 8; v++) {
        int grow = r0 + 64 * wm + 16 * mt + v + 8 * hi;
        if (grow < M) {
          float val = acc[mt][nt][v];
          if (bias) val += bias[gcol];
          if (Drow) val += Drow[(size_t)(grow & 63) * N + gcol];
          size_t off = (size_t)grow * N + gcol;
          if (accF) val += C[off];
          C[off] = val;
          if (Cbf) Cbf[off] = f2bf(val);
        }
      }
    }
  }
}

// ---------------------------------------------------------------------------
// Persistent GRU scan. 8 blocks; block k owns hidden units [64k,64k+64) i.e. the
// r/z/n gate columns {j, j+512, j+1024}. Wh slice (512x192 bf16, ~195 KB) stays
// resident in LDS; per step the shared h (64x512 bf16) is TDM-DMA'd into LDS
// (padded to a 520-ushort stride), gates computed with WMMA, then the GRU
// update runs elementwise. h is double-buffered in global bf16; one grid
// barrier per timestep.  LDS total ~309 KB of the 320 KB/WGP.
// ---------------------------------------------------------------------------
#define WHT_STR 520
#define HS_STR  520
#define GB_STR  196
#define WHT_BYTES (192 * WHT_STR * 2)
#define HS_BYTES  (64 * HS_STR * 2)
#define SCAN_LDS  (WHT_BYTES + HS_BYTES + 64 * GB_STR * 4)

__device__ __forceinline__ void gridbar(unsigned int* bar, unsigned int target) {
  __syncthreads();
  if (threadIdx.x == 0) {
    __threadfence();
    atomicAdd(bar, 1u);
    while (atomicAdd(bar, 0u) < target) { __builtin_amdgcn_s_sleep(1); }
    __threadfence();
#if defined(__has_builtin)
#if __has_builtin(__builtin_amdgcn_s_cluster_barrier)
    __builtin_amdgcn_s_cluster_barrier();   // NOP unless dispatched as a cluster
#endif
#endif
  }
  __syncthreads();
}

__global__ void __launch_bounds__(256)
mae_gru_scan(const float* __restrict__ xW,            // [T][64][1536]
             const unsigned short* __restrict__ Whbf, // [512][1536]
             const float* __restrict__ bh,            // [1536]
             float* __restrict__ hf32,                // [64][512] master state -> hT
             unsigned short* __restrict__ hbfA,       // [64][512] double buffer
             unsigned short* __restrict__ hbfB,
             float* __restrict__ ysf,                 // [T][64][512] or null
             unsigned short* __restrict__ ysbf,       // or null
             unsigned int* __restrict__ bar,
             int storeRev) {
  extern __shared__ char smem[];
  unsigned short* WhT = (unsigned short*)smem;                   // [192][WHT_STR]
  unsigned short* hS  = (unsigned short*)(smem + WHT_BYTES);     // [64][HS_STR]
  float* gbuf = (float*)(smem + WHT_BYTES + HS_BYTES);           // [64][GB_STR]
  const int tid = threadIdx.x, lane = tid & 31, wid = tid >> 5;
  const int chunk0 = blockIdx.x * 64;
  const int row16 = lane & 15, hi = lane >> 4;
  const int mt = wid & 3, nh = wid >> 2;                         // wave grid 4(M) x 2(96-col)
#if HAVE_TDM
  const unsigned int hs_base = (unsigned int)(size_t)(void*)hS;
#endif

  // Stage Wh slice transposed into LDS: WhT[localcol][k]
  for (int e = tid; e < 192 * 512; e += 256) {
    int j = e >> 9, k = e & 511;
    int gcol = (j >> 6) * 512 + chunk0 + (j & 63);               // r/z/n groups
    WhT[j * WHT_STR + k] = Whbf[(size_t)k * G3 + gcol];
  }
  // zero h (own chunk only) and the first bf16 read buffer
  for (int e = tid; e < 64 * 64; e += 256) {
    int m = e >> 6, j = e & 63;
    hf32[m * Hh + chunk0 + j] = 0.f;
    hbfA[m * Hh + chunk0 + j] = 0;
  }
  unsigned int phase = 1;
  gridbar(bar, NBLK * phase);

  const v8f vz = {0.f, 0.f, 0.f, 0.f, 0.f, 0.f, 0.f, 0.f};
  for (int t = 0; t < T_; t++) {
    const unsigned short* hread = (t & 1) ? hbfB : hbfA;
    unsigned short* hwrite = (t & 1) ? hbfA : hbfB;

    // DMA the full shared h (all 8 chunks) into LDS for this step's GEMM
#if HAVE_TDM
    if (wid == 0) {
      tdm_load_tile_2d(hs_base, hread, 512u, 64u, 512u, 512u, 64u, 7u, 3u);
      __builtin_amdgcn_s_wait_tensorcnt(0);
    }
#else
    for (int e = tid; e < 64 * 256; e += 256) {
      int m = e >> 8, kp = (e & 255) << 1;
      *(unsigned int*)&hS[m * HS_STR + kp] =
          *(const unsigned int*)(hread + (size_t)m * Hh + kp);
    }
#endif
    __syncthreads();

    // gates = h @ Wh_slice : M=64 (4 tiles), N=192 (12 tiles), K=512
    v8f acc[6];
    for (int i = 0; i < 6; i++) acc[i] = vz;
    for (int kk0 = 0; kk0 < Hh; kk0 += 32) {
      FragBF a;
      const unsigned short* hb = hS + (size_t)(16 * mt + row16) * HS_STR + kk0;
      for (int v = 0; v < 4; v++) {
        int k = 2 * v + 8 * hi;
        a.u[v]     = *(const unsigned int*)(hb + k);
        a.u[v + 4] = *(const unsigned int*)(hb + 16 + k);
      }
      for (int nt = 0; nt < 6; nt++) {
        FragBF b;
        const unsigned short* ws = WhT + (96 * nh + 16 * nt + row16) * WHT_STR + kk0;
        for (int v = 0; v < 8; v++)
          b.u[v] = *(const unsigned int*)(ws + 8 * hi + 2 * (v & 3) + 16 * (v >> 2));
        acc[nt] = __builtin_amdgcn_wmma_f32_16x16x32_bf16(
            false, a.v, false, b.v, (short)0, acc[nt], false, false);
      }
    }
    for (int nt = 0; nt < 6; nt++) {
      int lcol = 96 * nh + 16 * nt + row16;
      for (int v = 0; v < 8; v++)
        gbuf[(16 * mt + v + 8 * hi) * GB_STR + lcol] = acc[nt][v];
    }
    __syncthreads();

    // prefetch next step's xW slab (hidden behind this step's elementwise work)
    if (t + 1 < T_) {
      const char* pf = (const char*)(xW + (size_t)(t + 1) * Nn * G3) + (size_t)tid * 1536;
      for (int p = 0; p < 1536; p += 256) __builtin_prefetch(pf + p, 0, 1);
    }

    // elementwise GRU update on own 64x64 chunk
    const float* xwb = xW + (size_t)t * Nn * G3;
    const int tout = storeRev ? (T_ - 1 - t) : t;
    for (int e = tid; e < 64 * 64; e += 256) {
      int m = e >> 6, j = e & 63;
      int jg = chunk0 + j;
      float gr = gbuf[m * GB_STR + j]       + bh[jg];
      float gz = gbuf[m * GB_STR + 64 + j]  + bh[512 + jg];
      float gn = gbuf[m * GB_STR + 128 + j] + bh[1024 + jg];
      const float* xm = xwb + (size_t)m * G3;
      float r = 1.f / (1.f + __expf(-(xm[jg] + gr)));
      float z = 1.f / (1.f + __expf(-(xm[512 + jg] + gz)));
      float n = tanhf(xm[1024 + jg] + r * gn);
      float hnew = (1.f - z) * n + z * hf32[m * Hh + jg];
      hf32[m * Hh + jg] = hnew;
      hwrite[m * Hh + jg] = f2bf(hnew);
      if (ysf)  ysf[((size_t)tout * Nn + m) * Hh + jg] = hnew;
      if (ysbf) ysbf[((size_t)tout * Nn + m) * Hh + jg] = f2bf(hnew);
    }
    phase++;
    gridbar(bar, NBLK * phase);
  }
}

// ---------------------------------------------------------------------------
// msg = (sum over other agents)/7, emitted as bf16 (consumed only by a GEMM)
// ---------------------------------------------------------------------------
__global__ void mae_msg(const float* __restrict__ h, unsigned short* __restrict__ msgbf) {
  int i = blockIdx.x * blockDim.x + threadIdx.x;
  if (i >= Nn * Hh) return;
  int n = i >> 9, j = i & 511;
  int b = n >> 3;
  float s = 0.f;
  for (int a = 0; a < 8; a++) s += h[((b << 3) + a) * Hh + j];
  msgbf[i] = f2bf((s - h[n * Hh + j]) * (1.f / 7.f));
}

__global__ void mae_last_h(const float* __restrict__ h, float* __restrict__ out) {
  int i = blockIdx.x * blockDim.x + threadIdx.x;
  if (i >= 8 * Hh) return;
  int b = i >> 9, j = i & 511;
  out[i] = h[(b * 8) * Hh + j];   // agent 0 of each batch
}

// ---------------------------------------------------------------------------
// Host orchestration
// ---------------------------------------------------------------------------
extern "C" void kernel_launch(void* const* d_in, const int* in_sizes, int n_in,
                              void* d_out, int out_size, void* d_ws, size_t ws_size,
                              hipStream_t stream) {
  const float* article = (const float*)d_in[0];
  const float* Wx_f = (const float*)d_in[2];
  const float* Wh_f = (const float*)d_in[3];
  const float* bx_f = (const float*)d_in[4];
  const float* bh_f = (const float*)d_in[5];
  const float* Wx_b = (const float*)d_in[6];
  const float* Wh_b = (const float*)d_in[7];
  const float* bx_b = (const float*)d_in[8];
  const float* bh_b = (const float*)d_in[9];
  const float* Wp   = (const float*)d_in[10];
  const float* bp   = (const float*)d_in[11];
  const float* Wm   = (const float*)d_in[12];
  const float* bm   = (const float*)d_in[13];
  const float* Wx_c = (const float*)d_in[14];
  const float* Wh_c = (const float*)d_in[15];
  const float* bx_c = (const float*)d_in[16];
  const float* bh_c = (const float*)d_in[17];

  float* out_enc = (float*)d_out;
  float* out_lh  = out_enc + (size_t)T_ * Nn * Hh;

  char* wp_ = (char*)d_ws;
  auto alloc = [&](size_t b) -> void* {
    void* p = (void*)wp_;
    wp_ += (b + 255) & ~(size_t)255;
    return p;
  };

  const size_t ENC_EL = (size_t)T_ * Nn * Hh;        // 16.8M elements
  unsigned short* art_bf  = (unsigned short*)alloc(ENC_EL * 2);
  unsigned short* encf_bf = (unsigned short*)alloc(ENC_EL * 2);
  unsigned short* encb_bf = (unsigned short*)alloc(ENC_EL * 2);
  unsigned short* wxf_bf  = (unsigned short*)alloc((size_t)DIN_ * G3 * 2);
  unsigned short* whf_bf  = (unsigned short*)alloc((size_t)Hh * G3 * 2);
  unsigned short* wxb_bf  = (unsigned short*)alloc((size_t)DIN_ * G3 * 2);
  unsigned short* whb_bf  = (unsigned short*)alloc((size_t)Hh * G3 * 2);
  unsigned short* wp_bf   = (unsigned short*)alloc((size_t)1024 * Hh * 2);
  unsigned short* wm_bf   = (unsigned short*)alloc((size_t)1024 * Hh * 2);
  unsigned short* wxc_bf  = (unsigned short*)alloc((size_t)2 * Hh * G3 * 2);
  unsigned short* whc_bf  = (unsigned short*)alloc((size_t)2 * Hh * G3 * 2);
  float* xW    = (float*)alloc((size_t)T_ * Nn * G3 * 4);   // 201 MB
  float* C64   = (float*)alloc(ENC_EL * 4);                 // 67 MB f32 scratch
  float* hPf   = (float*)alloc((size_t)Nn * Hh * 4);
  float* msgrf = (float*)alloc((size_t)Nn * Hh * 4);
  float* h_f   = (float*)alloc((size_t)Nn * Hh * 4);
  float* h_b   = (float*)alloc((size_t)Nn * Hh * 4);
  float* h_L0  = (float*)alloc((size_t)Nn * Hh * 4);
  float* h_L1  = (float*)alloc((size_t)Nn * Hh * 4);
  unsigned short* hf_bf  = (unsigned short*)alloc((size_t)Nn * Hh * 2);
  unsigned short* hb_bf  = (unsigned short*)alloc((size_t)Nn * Hh * 2);
  unsigned short* msg_bf = (unsigned short*)alloc((size_t)Nn * Hh * 2);
  unsigned short* hbfA   = (unsigned short*)alloc((size_t)Nn * Hh * 2);
  unsigned short* hbfB   = (unsigned short*)alloc((size_t)Nn * Hh * 2);
  unsigned int* bar = (unsigned int*)alloc(256);
  // alias large bf16 buffers that are dead by the time these are written:
  unsigned short* encP_bf = art_bf;    // article dead after xW_b GEMM
  unsigned short* nxt_bf  = encf_bf;   // enc_f dead after projection pass 1
  unsigned short* encL_bf = encb_bf;   // enc_b dead after projection pass 2

  (void)hipFuncSetAttribute(reinterpret_cast<const void*>(&mae_gru_scan),
                            hipFuncAttributeMaxDynamicSharedMemorySize, SCAN_LDS);

  auto cvt = [&](const float* s, unsigned short* d, long long n) {
    int blocks = (int)((n + 255) / 256);
    if (blocks > 4096) blocks = 4096;
    mae_cvt_bf16<<<blocks, 256, 0, stream>>>(s, d, n);
  };
  auto gemm = [&](const unsigned short* A, int lda, const unsigned short* B, int ldb,
                  const float* bias, const float* Drow, float* C, unsigned short* Cbf,
                  int M, int N, int K, int accF, int revF) {
    dim3 g(N / 128, (M + 127) / 128);
    mae_gemm<<<g, 256, 0, stream>>>(A, lda, B, ldb, bias, Drow, C, Cbf, M, N, K, accF, revF);
  };
  auto scan = [&](const float* xw, const unsigned short* wh, const float* bh,
                  float* hstate, float* ysf, unsigned short* ysbf, int rev) {
    hipMemsetAsync(bar, 0, 256, stream);
    mae_gru_scan<<<NBLK, 256, SCAN_LDS, stream>>>(xw, wh, bh, hstate, hbfA, hbfB,
                                                  ysf, ysbf, bar, rev);
  };

  const int M = T_ * Nn;   // 32768

  // 1) bf16 conversions of activations and all weights
  cvt(article, art_bf, (long long)ENC_EL);
  cvt(Wx_f, wxf_bf, (long long)DIN_ * G3);
  cvt(Wh_f, whf_bf, (long long)Hh * G3);
  cvt(Wx_b, wxb_bf, (long long)DIN_ * G3);
  cvt(Wh_b, whb_bf, (long long)Hh * G3);
  cvt(Wp,   wp_bf,  (long long)1024 * Hh);
  cvt(Wm,   wm_bf,  (long long)1024 * Hh);
  cvt(Wx_c, wxc_bf, (long long)2 * Hh * G3);
  cvt(Wh_c, whc_bf, (long long)2 * Hh * G3);

  // 2) forward GRU: xW_f = article @ Wx_f + bx_f ; scan
  gemm(art_bf, DIN_, wxf_bf, G3, bx_f, nullptr, xW, nullptr, M, G3, DIN_, 0, 0);
  scan(xW, whf_bf, bh_f, h_f, nullptr, encf_bf, 0);
  cvt(h_f, hf_bf, (long long)Nn * Hh);

  // 3) backward GRU: xW_b from time-reversed article; outputs stored re-reversed
  gemm(art_bf, DIN_, wxb_bf, G3, bx_b, nullptr, xW, nullptr, M, G3, DIN_, 0, 1);
  scan(xW, whb_bf, bh_b, h_b, nullptr, encb_bf, 1);
  cvt(h_b, hb_bf, (long long)Nn * Hh);

  // 4) bidirectional projection: enc = [enc_f, enc_b] @ Wp + bp (split-K two-pass)
  gemm(encf_bf, Hh, wp_bf,                   Hh, bp, nullptr, C64, nullptr, M, Hh, Hh, 0, 0);
  gemm(encb_bf, Hh, wp_bf + (size_t)Hh * Hh, Hh, nullptr, nullptr, C64, encP_bf, M, Hh, Hh, 1, 0);
  // h = [h_f, h_b] @ Wp + bp
  gemm(hf_bf, Hh, wp_bf,                   Hh, bp, nullptr, hPf, nullptr, Nn, Hh, Hh, 0, 0);
  gemm(hb_bf, Hh, wp_bf + (size_t)Hh * Hh, Hh, nullptr, nullptr, hPf, nullptr, Nn, Hh, Hh, 1, 0);

  // 5) communication layers
  const unsigned short* encSrc = encP_bf;
  const float* hSrc = hPf;
  float* hDst[2] = {h_L0, h_L1};
  for (int l = 0; l < 2; l++) {
    mae_msg<<<(Nn * Hh + 255) / 256, 256, 0, stream>>>(hSrc, msg_bf);
    // msgrow = msg @ Wm_bot + bm  (broadcast over time in next GEMM)
    gemm(msg_bf, Hh, wm_bf + (size_t)Hh * Hh, Hh, bm, nullptr, msgrf, nullptr, Nn, Hh, Hh, 0, 0);
    // nxt = encSrc @ Wm_top + msgrow[row%64]
    gemm(encSrc, Hh, wm_bf, Hh, nullptr, msgrf, C64, nxt_bf, M, Hh, Hh, 0, 0);
    // xW_c = nxt @ Wx_c[l] + bx_c[l]
    gemm(nxt_bf, Hh, wxc_bf + (size_t)l * Hh * G3, G3, bx_c + (size_t)l * G3,
         nullptr, xW, nullptr, M, G3, Hh, 0, 0);
    // GRU scan; final layer writes f32 enc straight into d_out
    scan(xW, whc_bf + (size_t)l * Hh * G3, bh_c + (size_t)l * G3, hDst[l],
         (l == 1) ? out_enc : nullptr, (l == 0) ? encL_bf : nullptr, 0);
    encSrc = encL_bf;
    hSrc = hDst[l];
  }

  // 6) last hidden of agent 0 per batch
  mae_last_h<<<(8 * Hh + 255) / 256, 256, 0, stream>>>(h_L1, out_lh);
}
